// MinkNeXtBlock_54898271977488
// MI455X (gfx1250) — matrix operand order repaired and checked by reading, not compiled, not run on hardware
//
#include <hip/hip_runtime.h>
#include <hip/hip_bf16.h>

// ---------------------------------------------------------------------------
// MinkNeXt block for MI455X (gfx1250, wave32, WMMA f32<-f16 16x16x32)
// ---------------------------------------------------------------------------

typedef __attribute__((ext_vector_type(16))) _Float16 v16h;
typedef __attribute__((ext_vector_type(8)))  _Float16 v8h;
typedef __attribute__((ext_vector_type(8)))  float    v8f;
typedef __attribute__((ext_vector_type(4)))  float    v4f;

#define NPTS   100000
#define NK     27
#define NTILES 6250          // NPTS / 16 exactly
#define BLOCKS 391           // persistent-ish: 391*8 = 3128 waves, 2 tiles/wave
#define TSTRIDE (BLOCKS * 8)

// workspace layout (bytes)
#define OFF_H0 0u                       // [N,32]  f16 : 6,400,000
#define OFF_H1 6400000u                 // [N,128] f16 : 25,600,000
#define OFF_WA 32000000u                // swizzled W_conv f16 : 2048
#define OFF_W1 32002048u                // swizzled W1 f16 : 221,184
#define OFF_W2 32223232u                // swizzled W2 f16 : 221,184

__device__ __forceinline__ float gelu_erf(float v) {
    return 0.5f * v * (1.0f + erff(v * 0.70710678118654752440f));
}

__device__ __forceinline__ v16h join8(v8h lo, v8h hi) {
    return __builtin_shufflevector(lo, hi, 0,1,2,3,4,5,6,7,8,9,10,11,12,13,14,15);
}

// Butterfly add via ds_swizzle_b32 (group-of-32 XOR mode: and=0x1f, or=0, xor=M).
// No clamp/address VALU, unlike generic __shfl_xor.
template <int M>
__device__ __forceinline__ float swadd(float v) {
    int t = __builtin_amdgcn_ds_swizzle(__float_as_int(v), (M << 10) | 0x1f);
    return v + __int_as_float(t);
}
__device__ __forceinline__ float red16(float v) {   // sum over 16-lane group
    return swadd<8>(swadd<4>(swadd<2>(swadd<1>(v))));
}

// ---------------------------------------------------------------------------
// Weight prep: f32 -> f16, swizzled into CDNA5 B-fragment per-lane layout.
// B fragment (16-bit, 32x16): lane = hi*16+n ; element e holds B[16*(e>>3)+8*hi+(e&7)][16*t+n]
// ---------------------------------------------------------------------------
__global__ void __launch_bounds__(256) prep_weights(
    const float* __restrict__ Wc,   // [32,32]
    const float* __restrict__ W1,   // [27,32,128]
    const float* __restrict__ W2,   // [27,128,32]
    _Float16* __restrict__ wa, _Float16* __restrict__ w1s, _Float16* __restrict__ w2s)
{
    int g = blockIdx.x * blockDim.x + threadIdx.x;     // fragment-lane chunk id
    if (g < 64) {                                      // W_conv: t(2) x lane(32)
        int t = g >> 5, lane = g & 31, n = lane & 15, hi = lane >> 4;
        #pragma unroll
        for (int e = 0; e < 16; ++e)
            wa[g * 16 + e] = (_Float16)Wc[(16 * (e >> 3) + 8 * hi + (e & 7)) * 32 + 16 * t + n];
    } else if (g < 64 + 6912) {                        // W1: (k*8+t)*32+lane
        int c = g - 64;
        int lane = c & 31, t = (c >> 5) & 7, k = c >> 8;
        int n = lane & 15, hi = lane >> 4;
        #pragma unroll
        for (int e = 0; e < 16; ++e)
            w1s[c * 16 + e] = (_Float16)W1[(k * 32 + 16 * (e >> 3) + 8 * hi + (e & 7)) * 128 + 16 * t + n];
    } else if (g < 64 + 6912 + 6912) {                 // W2: ((k*4+kk)*2+t)*32+lane
        int c = g - 64 - 6912;
        int lane = c & 31, t = (c >> 5) & 1, kk = (c >> 6) & 3, k = c >> 8;
        int n = lane & 15, hi = lane >> 4;
        #pragma unroll
        for (int e = 0; e < 16; ++e)
            w2s[c * 16 + e] = (_Float16)W2[(k * 128 + kk * 32 + 16 * (e >> 3) + 8 * hi + (e & 7)) * 32 + 16 * t + n];
    }
}

// ---------------------------------------------------------------------------
// Stage A: h0 = f16( gelu( LN( x @ W_conv ) ) )       one 16-pt tile per wave
// ---------------------------------------------------------------------------
__global__ void __launch_bounds__(256) stageA(
    const float* __restrict__ x, const float* __restrict__ gamma0,
    const float* __restrict__ beta0, const _Float16* __restrict__ waG,
    _Float16* __restrict__ h0)
{
    extern __shared__ char smem[];
    _Float16* wa  = (_Float16*)smem;            // 2 KB
    _Float16* stg = (_Float16*)(smem + 2048);   // 8 waves * 1 KB

    int tid = threadIdx.x;
    if (tid < 128) ((float4*)wa)[tid] = ((const float4*)waG)[tid];
    __syncthreads();

    int wave = tid >> 5, lane = tid & 31, n = lane & 15, hi = lane >> 4;
    float g[2]  = { gamma0[n], gamma0[16 + n] };
    float bt[2] = { beta0[n],  beta0[16 + n]  };
    v16h b0 = *(const v16h*)(wa + (0 * 32 + lane) * 16);
    v16h b1 = *(const v16h*)(wa + (1 * 32 + lane) * 16);
    _Float16* st = stg + wave * 512;

    for (int tile = blockIdx.x * 8 + wave; tile < NTILES; tile += TSTRIDE) {
        int base = tile * 16;

        // A fragment: row m = n, channels {8hi..8hi+7, 16+8hi..16+8hi+7}
        const float* xr = x + (size_t)(base + n) * 32;
        v4f c0 = *(const v4f*)(xr + 8 * hi);
        v4f c1 = *(const v4f*)(xr + 8 * hi + 4);
        v4f c2 = *(const v4f*)(xr + 16 + 8 * hi);
        v4f c3 = *(const v4f*)(xr + 16 + 8 * hi + 4);
        v16h a;
        #pragma unroll
        for (int e = 0; e < 4; ++e) {
            a[e]      = (_Float16)c0[e]; a[4 + e]  = (_Float16)c1[e];
            a[8 + e]  = (_Float16)c2[e]; a[12 + e] = (_Float16)c3[e];
        }

        v8f acc0 = {}, acc1 = {};
        acc0 = __builtin_amdgcn_wmma_f32_16x16x32_f16(false, a, false, b0, (short)0, acc0, false, false);
        acc1 = __builtin_amdgcn_wmma_f32_16x16x32_f16(false, a, false, b1, (short)0, acc1, false, false);

        #pragma unroll
        for (int r = 0; r < 8; ++r) {
            float v0 = acc0[r], v1 = acc1[r];
            float s = red16(v0 + v1);
            float q = red16(v0 * v0 + v1 * v1);
            float mu = s * (1.0f / 32.0f);
            float rs = rsqrtf(q * (1.0f / 32.0f) - mu * mu + 1e-6f);
            int m = r + 8 * hi;
            st[m * 32 + n]      = (_Float16)gelu_erf((v0 - mu) * rs * g[0] + bt[0]);
            st[m * 32 + 16 + n] = (_Float16)gelu_erf((v1 - mu) * rs * g[1] + bt[1]);
        }
        // coalesced writeback of the 1 KB tile
        #pragma unroll
        for (int j = 0; j < 2; ++j)
            ((float4*)(h0 + (size_t)base * 32))[lane * 2 + j] = ((const float4*)st)[lane * 2 + j];
    }
}

// ---------------------------------------------------------------------------
// Stage B: h1 = f16( gelu( LN( sum_k mask * h0[idx] @ W1[k] ) ) )   [N,128]
// ---------------------------------------------------------------------------
__global__ void __launch_bounds__(256) stageB(
    const _Float16* __restrict__ h0, const int* __restrict__ nidx,
    const unsigned char* __restrict__ nmask, const float* __restrict__ gamma1,
    const float* __restrict__ beta1, const _Float16* __restrict__ w1G,
    _Float16* __restrict__ h1)
{
    extern __shared__ char smem[];
    _Float16* w   = (_Float16*)smem;              // 221,184 B swizzled W1
    _Float16* stg = (_Float16*)(smem + 221184);   // 8 waves * 4 KB staging

    int tid = threadIdx.x;
    for (int i = tid; i < 13824; i += 256)        // 221184 B / 16
        ((float4*)w)[i] = ((const float4*)w1G)[i];
    __syncthreads();

    int wave = tid >> 5, lane = tid & 31, n = lane & 15, hi = lane >> 4;
    float g[8], bt[8];
    #pragma unroll
    for (int t = 0; t < 8; ++t) { g[t] = gamma1[t * 16 + n]; bt[t] = beta1[t * 16 + n]; }
    _Float16* st = stg + wave * 2048;

    for (int tile = blockIdx.x * 8 + wave; tile < NTILES; tile += TSTRIDE) {
        int base = tile * 16;
        int p = base + n;
        const int* ip = nidx + (size_t)p * NK;
        const unsigned char* mp = nmask + (size_t)p * NK;

        v8f acc[8];
        #pragma unroll
        for (int t = 0; t < 8; ++t) acc[t] = (v8f){};

        for (int k = 0; k < NK; ++k) {
            int idx = ip[k];
            v16h a;
            if (mp[k]) {
                const _Float16* row = h0 + (size_t)idx * 32;
                a = join8(*(const v8h*)(row + 8 * hi), *(const v8h*)(row + 16 + 8 * hi));
            } else {
                a = (v16h){};
            }
            #pragma unroll
            for (int t = 0; t < 8; ++t) {
                v16h b = *(const v16h*)(w + ((k * 8 + t) * 32 + lane) * 16);
                acc[t] = __builtin_amdgcn_wmma_f32_16x16x32_f16(false, a, false, b, (short)0, acc[t], false, false);
            }
        }

        #pragma unroll
        for (int r = 0; r < 8; ++r) {
            float s = 0.0f, q = 0.0f;
            #pragma unroll
            for (int t = 0; t < 8; ++t) { float v = acc[t][r]; s += v; q += v * v; }
            s = red16(s);
            q = red16(q);
            float mu = s * (1.0f / 128.0f);
            float rs = rsqrtf(q * (1.0f / 128.0f) - mu * mu + 1e-6f);
            int m = r + 8 * hi;
            #pragma unroll
            for (int t = 0; t < 8; ++t)
                st[m * 128 + t * 16 + n] = (_Float16)gelu_erf((acc[t][r] - mu) * rs * g[t] + bt[t]);
        }
        // coalesced writeback of the 4 KB tile
        #pragma unroll
        for (int j = 0; j < 8; ++j)
            ((float4*)(h1 + (size_t)base * 128))[lane * 8 + j] = ((const float4*)st)[lane * 8 + j];
    }
}

// ---------------------------------------------------------------------------
// Stage C: out = gelu( LN( sum_k mask * h1[idx] @ W2[k] ) + x )     [N,32] f32
// ---------------------------------------------------------------------------
__global__ void __launch_bounds__(256) stageC(
    const _Float16* __restrict__ h1, const int* __restrict__ nidx,
    const unsigned char* __restrict__ nmask, const float* __restrict__ gamma2,
    const float* __restrict__ beta2, const _Float16* __restrict__ w2G,
    const float* __restrict__ x, float* __restrict__ out)
{
    extern __shared__ char smem[];
    _Float16* w  = (_Float16*)smem;             // 221,184 B swizzled W2
    float*    stg = (float*)(smem + 221184);    // 8 waves * 2 KB staging

    int tid = threadIdx.x;
    for (int i = tid; i < 13824; i += 256)
        ((float4*)w)[i] = ((const float4*)w2G)[i];
    __syncthreads();

    int wave = tid >> 5, lane = tid & 31, n = lane & 15, hi = lane >> 4;
    float g[2]  = { gamma2[n], gamma2[16 + n] };
    float bt[2] = { beta2[n],  beta2[16 + n]  };
    float* st = stg + wave * 512;

    for (int tile = blockIdx.x * 8 + wave; tile < NTILES; tile += TSTRIDE) {
        int base = tile * 16;
        int p = base + n;
        const int* ip = nidx + (size_t)p * NK;
        const unsigned char* mp = nmask + (size_t)p * NK;

        v8f acc0 = {}, acc1 = {};

        for (int k = 0; k < NK; ++k) {
            int idx = ip[k];
            bool msk = mp[k] != 0;
            const _Float16* row = h1 + (size_t)idx * 128;
            #pragma unroll
            for (int kk = 0; kk < 4; ++kk) {
                v16h a;
                if (msk)
                    a = join8(*(const v8h*)(row + kk * 32 + 8 * hi),
                              *(const v8h*)(row + kk * 32 + 16 + 8 * hi));
                else
                    a = (v16h){};
                v16h b0 = *(const v16h*)(w + (((k * 4 + kk) * 2 + 0) * 32 + lane) * 16);
                v16h b1 = *(const v16h*)(w + (((k * 4 + kk) * 2 + 1) * 32 + lane) * 16);
                acc0 = __builtin_amdgcn_wmma_f32_16x16x32_f16(false, a, false, b0, (short)0, acc0, false, false);
                acc1 = __builtin_amdgcn_wmma_f32_16x16x32_f16(false, a, false, b1, (short)0, acc1, false, false);
            }
        }

        #pragma unroll
        for (int r = 0; r < 8; ++r) {
            float v0 = acc0[r], v1 = acc1[r];
            float s = red16(v0 + v1);
            float q = red16(v0 * v0 + v1 * v1);
            float mu = s * (1.0f / 32.0f);
            float rs = rsqrtf(q * (1.0f / 32.0f) - mu * mu + 1e-6f);
            int m = r + 8 * hi;
            st[m * 32 + n]      = (v0 - mu) * rs * g[0] + bt[0];
            st[m * 32 + 16 + n] = (v1 - mu) * rs * g[1] + bt[1];
        }
        // coalesced writeback + residual + gelu: 2 KB tile, 64 B per lane
        #pragma unroll
        for (int j = 0; j < 4; ++j) {
            float4 v  = ((const float4*)st)[lane * 4 + j];
            float4 rx = ((const float4*)(x + (size_t)base * 32))[lane * 4 + j];
            float4 o;
            o.x = gelu_erf(v.x + rx.x);
            o.y = gelu_erf(v.y + rx.y);
            o.z = gelu_erf(v.z + rx.z);
            o.w = gelu_erf(v.w + rx.w);
            ((float4*)(out + (size_t)base * 32))[lane * 4 + j] = o;
        }
    }
}

// ---------------------------------------------------------------------------
extern "C" void kernel_launch(void* const* d_in, const int* in_sizes, int n_in,
                              void* d_out, int out_size, void* d_ws, size_t ws_size,
                              hipStream_t stream)
{
    (void)in_sizes; (void)n_in; (void)out_size; (void)ws_size;
    const float*         x     = (const float*)d_in[0];
    const int*           nidx  = (const int*)d_in[1];
    const unsigned char* nmask = (const unsigned char*)d_in[2];   // numpy bool: 1 byte
    const float*         Wc    = (const float*)d_in[3];
    const float*         g0    = (const float*)d_in[4];
    const float*         b0    = (const float*)d_in[5];
    const float*         W1    = (const float*)d_in[6];
    const float*         g1    = (const float*)d_in[7];
    const float*         b1    = (const float*)d_in[8];
    const float*         W2    = (const float*)d_in[9];
    const float*         g2    = (const float*)d_in[10];
    const float*         b2    = (const float*)d_in[11];
    float* out = (float*)d_out;

    char* ws = (char*)d_ws;
    _Float16* h0  = (_Float16*)(ws + OFF_H0);
    _Float16* h1  = (_Float16*)(ws + OFF_H1);
    _Float16* wa  = (_Float16*)(ws + OFF_WA);
    _Float16* w1s = (_Float16*)(ws + OFF_W1);
    _Float16* w2s = (_Float16*)(ws + OFF_W2);

    prep_weights<<<55, 256, 0, stream>>>(Wc, W1, W2, wa, w1s, w2s);
    stageA<<<BLOCKS, 256, 2048 + 8 * 1024, stream>>>(x, g0, b0, wa, h0);
    stageB<<<BLOCKS, 256, 221184 + 8 * 4096, stream>>>(h0, nidx, nmask, g1, b1, w1s, h1);
    stageC<<<BLOCKS, 256, 221184 + 8 * 2048, stream>>>(h1, nidx, nmask, g2, b2, w2s, x, out);
}
